// LSTMDecoder_43293270344053
// MI455X (gfx1250) — compile-verified, matrix operand order checked by reading, not run on hardware
//
#include <hip/hip_runtime.h>

// ---------------- problem constants ----------------
constexpr int kB   = 8;
constexpr int kS   = 256;
constexpr int kSKV = 1024;
constexpr int kH   = 512;
constexpr int kNH  = 8;
constexpr int kHD  = 64;      // kH / kNH
constexpr int kV   = 32000;
constexpr int kG4  = 2048;    // 4*kH
constexpr int kBS  = kB * kS; // 2048 decode positions
constexpr int kBKV = kB * kSKV;

typedef __attribute__((ext_vector_type(16))) __bf16 v16bf;
typedef __attribute__((ext_vector_type(8)))  float  v8f;

union Frag {
  v16bf v;
  unsigned int u[8];
};

__device__ __forceinline__ unsigned short f2bf(float f) {
  unsigned int u = __float_as_uint(f);
  unsigned int r = (u + 0x7FFFu + ((u >> 16) & 1u)) >> 16;  // RNE
  return (unsigned short)r;
}

// ---------------- fp32 -> bf16 conversion ----------------
__global__ void cvt_bf16_kernel(const float* __restrict__ src,
                                unsigned short* __restrict__ dst, int n) {
  int i = blockIdx.x * blockDim.x + threadIdx.x;
  int stride = gridDim.x * blockDim.x;
  for (; i < n; i += stride) dst[i] = f2bf(src[i]);
}

// ---------------- bias fold: b_ih + b_hh ----------------
__global__ void bsum_kernel(const float* __restrict__ a,
                            const float* __restrict__ b,
                            float* __restrict__ c, int n) {
  int i = blockIdx.x * blockDim.x + threadIdx.x;
  if (i < n) c[i] = a[i] + b[i];
}

// ---------------- embedding gather (bf16 rows) ----------------
// one block per token position, 256 threads copy 256 dwords (512 bf16)
__global__ void gather_kernel(const int* __restrict__ toks,
                              const unsigned short* __restrict__ emb_bf,
                              unsigned short* __restrict__ x_bf) {
  int r = blockIdx.x;
  int tok = toks[r];
  const unsigned int* s = (const unsigned int*)(emb_bf + (size_t)tok * kH);
  unsigned int* d = (unsigned int*)(x_bf + (size_t)r * kH);
  d[threadIdx.x] = s[threadIdx.x];
}

// ---------------- tiled WMMA GEMM: C[M,N] = A[M,K] * W[N,K]^T + bias[n] ----
// All matrices row-major. A, W in bf16; C fp32. M%64==0, N%256==0, K%32==0.
// block = 256 threads (8 waves) arranged 2(M) x 4(N); each wave owns a
// 32x64 micro-tile (2 A-frags x 4 B-frags -> 8 WMMAs per K-step) for high
// register-level reuse: 8 wmma per 12 b128 loads.
__global__ void __launch_bounds__(256)
gemm_bf16_bt(const unsigned short* __restrict__ A,
             const unsigned short* __restrict__ Bw,
             const float* __restrict__ bias,
             float* __restrict__ C, int M, int N, int K) {
  int lane = threadIdx.x & 31;
  int wave = threadIdx.x >> 5;                 // 0..7
  int m0 = blockIdx.y * 64  + (wave >> 2) * 32;  // wave covers rows m0..m0+31
  int n0 = blockIdx.x * 256 + (wave & 3) * 64;   // wave covers cols n0..n0+63
  int grp = lane >> 4;                         // K-half select (ISA A/B layout)
  int lr  = lane & 15;                         // row (A) / col (B)

  const unsigned short* arow[2];
  const unsigned short* brow[4];
#pragma unroll
  for (int mi = 0; mi < 2; ++mi) arow[mi] = A + (size_t)(m0 + mi * 16 + lr) * K;
#pragma unroll
  for (int ni = 0; ni < 4; ++ni) brow[ni] = Bw + (size_t)(n0 + ni * 16 + lr) * K;

  v8f acc[2][4];
#pragma unroll
  for (int mi = 0; mi < 2; ++mi)
#pragma unroll
    for (int ni = 0; ni < 4; ++ni) acc[mi][ni] = (v8f){};

  for (int kk = 0; kk < K; kk += 32) {
    Frag af[2], bf[4];
#pragma unroll
    for (int i = 0; i < 8; ++i) {
      // 16-bit A/B 16x32 layout: vgpr i holds a K-pair; grp selects the +8
      // window, i>=4 selects the +16 window. Per-lane bytes are contiguous
      // in groups of 16B -> compiler fuses into global_load_b128.
      int k = kk + ((i & 4) << 2) + ((i & 3) << 1) + (grp << 3);
#pragma unroll
      for (int mi = 0; mi < 2; ++mi) af[mi].u[i] = *(const unsigned int*)(arow[mi] + k);
#pragma unroll
      for (int ni = 0; ni < 4; ++ni) bf[ni].u[i] = *(const unsigned int*)(brow[ni] + k);
    }
#pragma unroll
    for (int mi = 0; mi < 2; ++mi)
#pragma unroll
      for (int ni = 0; ni < 4; ++ni)
        acc[mi][ni] = __builtin_amdgcn_wmma_f32_16x16x32_bf16(
            false, af[mi].v, false, bf[ni].v, (short)0, acc[mi][ni], false, false);
  }

#pragma unroll
  for (int ni = 0; ni < 4; ++ni) {
    float bb = bias ? bias[n0 + ni * 16 + lr] : 0.0f;
#pragma unroll
    for (int mi = 0; mi < 2; ++mi) {
#pragma unroll
      for (int i = 0; i < 8; ++i) {
        // D layout: vgpr i -> M = i + 8*grp ; lane%16 -> N
        int m = m0 + mi * 16 + grp * 8 + i;
        C[(size_t)m * N + (n0 + ni * 16 + lr)] = acc[mi][ni][i] + bb;
      }
    }
  }
}

// ---------------- sequential scan: one block per (batch, head) -------------
// Each block replays the full LSTM recurrence for its batch (M=1 WMMA GEMV
// against bf16 W_hh) and computes only its head's attention slice. Blocks are
// fully independent -> no atomics, deterministic, graph-capture safe.
__global__ void __launch_bounds__(256)
scan_kernel(const float* __restrict__ xW,              // [kBS, kG4], biases folded
            const unsigned short* __restrict__ Whh_bf, // [kG4, kH]
            const float* __restrict__ Wq,              // [kH, kH]
            const float* __restrict__ bq,              // [kH]
            const float* __restrict__ Kmat,            // [kBKV, kH] fp32
            const float* __restrict__ Vmat,            // [kBKV, kH] fp32
            const int* __restrict__ mask,              // [kBKV]
            const float* __restrict__ h0,              // [kB, kH]
            const float* __restrict__ c0,              // [kB, kH]
            unsigned short* __restrict__ ctx_bf) {     // [kBS, kH]
  __shared__ float hS[kH];
  __shared__ float cS[kH];
  __shared__ unsigned short hbf[kH];
  __shared__ float gates[kG4];
  __shared__ float qS[kHD];
  __shared__ float sc[kSKV];
  __shared__ float red[256];
  __shared__ float part[4][kHD];

  int b  = blockIdx.x >> 3;   // batch
  int hd = blockIdx.x & 7;    // head
  int tid = threadIdx.x;
  int lane = tid & 31, wave = tid >> 5;
  int grp = lane >> 4, lr = lane & 15;

  for (int j = tid; j < kH; j += 256) {
    float hv = h0[b * kH + j];
    hS[j] = hv;
    cS[j] = c0[b * kH + j];
    hbf[j] = f2bf(hv);
  }
  __syncthreads();

  for (int t = 0; t < kS; ++t) {
    const float* xw = xW + (size_t)(b * kS + t) * kG4;

    // ---- gates = h @ W_hh^T (M=1 WMMA) + xw ----
    for (int nt = wave * 16; nt < (wave + 1) * 16; ++nt) {  // 128 N-tiles / 8 waves
      v8f acc = {};
      const unsigned short* brow = Whh_bf + (size_t)(nt * 16 + lr) * kH;
      for (int kk = 0; kk < kH; kk += 32) {
        Frag a, bb;
#pragma unroll
        for (int i = 0; i < 8; ++i) {
          int k = kk + ((i & 4) << 2) + ((i & 3) << 1) + (grp << 3);
          a.u[i]  = (lr == 0) ? *(const unsigned int*)&hbf[k] : 0u;  // row 0 only
          bb.u[i] = *(const unsigned int*)(brow + k);
        }
        acc = __builtin_amdgcn_wmma_f32_16x16x32_bf16(false, a.v, false, bb.v,
                                                      (short)0, acc, false, false);
      }
      if (lane < 16) gates[nt * 16 + lane] = acc[0] + xw[nt * 16 + lane];
    }
    __syncthreads();

    // ---- LSTM elementwise (PyTorch gate order i,f,g,o) ----
    for (int j = tid; j < kH; j += 256) {
      float ig = gates[j];
      float fg = gates[kH + j];
      float gg = gates[2 * kH + j];
      float og = gates[3 * kH + j];
      float si = 1.0f / (1.0f + __expf(-ig));
      float sf = 1.0f / (1.0f + __expf(-fg));
      float so = 1.0f / (1.0f + __expf(-og));
      float cn = sf * cS[j] + si * tanhf(gg);
      float hn = so * tanhf(cn);
      cS[j] = cn;
      hS[j] = hn;
      hbf[j] = f2bf(hn);
    }
    __syncthreads();

    // ---- q slice for this head ----
    if (tid < kHD) {
      int n = hd * kHD + tid;
      const float* wr = Wq + (size_t)n * kH;
      float s = bq[n];
      for (int j = 0; j < kH; ++j) s += hS[j] * wr[j];
      qS[tid] = s;
    }
    __syncthreads();

    // ---- scores ----
    for (int k = tid; k < kSKV; k += 256) {
      const float* kr = Kmat + (size_t)(b * kSKV + k) * kH + hd * kHD;
      float s = 0.0f;
#pragma unroll 8
      for (int d = 0; d < kHD; ++d) s += qS[d] * kr[d];
      s *= 0.125f;  // 1/sqrt(64)
      if (mask[b * kSKV + k] != 1) s = -1e9f;
      sc[k] = s;
    }
    __syncthreads();

    // ---- softmax (max, exp, sum) ----
    float lm = -3.4e38f;
    for (int k = tid; k < kSKV; k += 256) lm = fmaxf(lm, sc[k]);
    red[tid] = lm;
    __syncthreads();
    for (int off = 128; off > 0; off >>= 1) {
      if (tid < off) red[tid] = fmaxf(red[tid], red[tid + off]);
      __syncthreads();
    }
    float mx = red[0];
    __syncthreads();
    float ls = 0.0f;
    for (int k = tid; k < kSKV; k += 256) {
      float e = __expf(sc[k] - mx);
      sc[k] = e;
      ls += e;
    }
    red[tid] = ls;
    __syncthreads();
    for (int off = 128; off > 0; off >>= 1) {
      if (tid < off) red[tid] += red[tid + off];
      __syncthreads();
    }
    float inv = 1.0f / red[0];
    __syncthreads();

    // ---- ctx = attn @ V slice; write bf16 ----
    {
      int g = tid >> 6;   // 0..3 k-quarter
      int d = tid & 63;
      const float* vb = Vmat + (size_t)(b * kSKV + g * 256) * kH + hd * kHD + d;
      float s = 0.0f;
      for (int k = 0; k < 256; ++k) s += sc[g * 256 + k] * vb[(size_t)k * kH];
      part[g][d] = s;
      __syncthreads();
      if (tid < kHD) {
        float v = (part[0][tid] + part[1][tid] + part[2][tid] + part[3][tid]) * inv;
        ctx_bf[(size_t)(b * kS + t) * kH + hd * kHD + tid] = f2bf(v);
      }
    }
    __syncthreads();  // protect sc/part/gates for next step
  }
}

// ---------------- host-side orchestration ----------------
extern "C" void kernel_launch(void* const* d_in, const int* in_sizes, int n_in,
                              void* d_out, int out_size, void* d_ws, size_t ws_size,
                              hipStream_t stream) {
  (void)in_sizes; (void)n_in; (void)out_size; (void)ws_size;

  const int*   toks = (const int*)d_in[0];
  const float* enc  = (const float*)d_in[1];
  const int*   mask = (const int*)d_in[2];
  const float* emb  = (const float*)d_in[3];
  const float* W_ih = (const float*)d_in[4];
  const float* W_hh = (const float*)d_in[5];
  const float* b_ih = (const float*)d_in[6];
  const float* b_hh = (const float*)d_in[7];
  const float* Wq   = (const float*)d_in[8];
  const float* bq   = (const float*)d_in[9];
  const float* Wk   = (const float*)d_in[10];
  const float* bk   = (const float*)d_in[11];
  const float* Wv   = (const float*)d_in[12];
  const float* bv   = (const float*)d_in[13];
  const float* Wo   = (const float*)d_in[14];
  const float* bo   = (const float*)d_in[15];
  const float* h0   = (const float*)d_in[16];
  const float* c0   = (const float*)d_in[17];
  float* out = (float*)d_out;

  // workspace carve-up (~108 MB total)
  char* p = (char*)d_ws;
  auto alloc = [&](size_t bytes) -> char* {
    char* r = p;
    p += (bytes + 255) & ~(size_t)255;
    return r;
  };
  unsigned short* emb_bf  = (unsigned short*)alloc((size_t)kV  * kH * 2);
  unsigned short* enc_bf  = (unsigned short*)alloc((size_t)kBKV * kH * 2);
  unsigned short* Wih_bf  = (unsigned short*)alloc((size_t)kG4 * kH * 2);
  unsigned short* Whh_bf  = (unsigned short*)alloc((size_t)kG4 * kH * 2);
  unsigned short* Wk_bf   = (unsigned short*)alloc((size_t)kH * kH * 2);
  unsigned short* Wv_bf   = (unsigned short*)alloc((size_t)kH * kH * 2);
  unsigned short* Wo_bf   = (unsigned short*)alloc((size_t)kH * kH * 2);
  unsigned short* x_bf    = (unsigned short*)alloc((size_t)kBS * kH * 2);
  float*          bsum    = (float*)alloc((size_t)kG4 * 4);
  float*          xW      = (float*)alloc((size_t)kBS * kG4 * 4);
  float*          Kmat    = (float*)alloc((size_t)kBKV * kH * 4);
  float*          Vmat    = (float*)alloc((size_t)kBKV * kH * 4);
  unsigned short* ctx_bf  = (unsigned short*)alloc((size_t)kBS * kH * 2);
  float*          outs    = (float*)alloc((size_t)kBS * kH * 4);
  unsigned short* outs_bf = (unsigned short*)alloc((size_t)kBS * kH * 2);

  const int T = 256;
  auto cvt = [&](const float* s, unsigned short* d, int n) {
    cvt_bf16_kernel<<<(n + T - 1) / T, T, 0, stream>>>(s, d, n);
  };

  // phase 0: precision prep
  cvt(emb,  emb_bf,  kV * kH);
  cvt(enc,  enc_bf,  kBKV * kH);
  cvt(W_ih, Wih_bf,  kG4 * kH);
  cvt(W_hh, Whh_bf,  kG4 * kH);
  cvt(Wk,   Wk_bf,   kH * kH);
  cvt(Wv,   Wv_bf,   kH * kH);
  cvt(Wo,   Wo_bf,   kH * kH);
  bsum_kernel<<<kG4 / T, T, 0, stream>>>(b_ih, b_hh, bsum, kG4);
  gather_kernel<<<kBS, T, 0, stream>>>(toks, emb_bf, x_bf);

  // phase 1: hoisted projections (all WMMA). grid = (N/256, M/64)
  gemm_bf16_bt<<<dim3(kH / 256, kBKV / 64), T, 0, stream>>>(enc_bf, Wk_bf, bk, Kmat, kBKV, kH, kH);
  gemm_bf16_bt<<<dim3(kH / 256, kBKV / 64), T, 0, stream>>>(enc_bf, Wv_bf, bv, Vmat, kBKV, kH, kH);
  gemm_bf16_bt<<<dim3(kG4 / 256, kBS / 64), T, 0, stream>>>(x_bf, Wih_bf, bsum, xW, kBS, kG4, kH);

  // phase 2: sequential scan, 64 independent (batch, head) blocks
  scan_kernel<<<kB * kNH, T, 0, stream>>>(xW, Whh_bf, Wq, bq, Kmat, Vmat, mask,
                                          h0, c0, ctx_bf);

  // phase 3: output projection + tied logits (WMMA, fp32 out -> d_out)
  gemm_bf16_bt<<<dim3(kH / 256, kBS / 64), T, 0, stream>>>(ctx_bf, Wo_bf, bo, outs, kBS, kH, kH);
  cvt(outs, outs_bf, kBS * kH);
  gemm_bf16_bt<<<dim3(kV / 256, kBS / 64), T, 0, stream>>>(outs_bf, emb_bf, nullptr, out, kBS, kV, kH);
}
